// Attention_10015863734711
// MI455X (gfx1250) — matrix operand order repaired and checked by reading, not compiled, not run on hardware
//
#include <hip/hip_runtime.h>

// ---------------------------------------------------------------------------
// Relative-position causal attention for MI455X (gfx1250, wave32, WMMA).
// B=8, L=2048, D=64, REL=33, CLIP=16.
//
// bf16 flash attention, one wave32 per 16-row Q tile.
//   ws layout: Qbf[B*L*64] bf16 | Kbf[B*L*64] bf16 | Vt[B][64][L] bf16 |
//              drel[B*L*33] f32 | vrelT[64][32] bf16 (zero-padded t>16)
//
// Scheduling notes (v2):
//  - V B-fragments loaded into dedicated regs right after the S WMMAs, so the
//    softmax VALU block hides their global-load latency before the PV WMMAs.
//  - value_relative B-fragments are loop-invariant -> hoisted out of kv loop.
// ---------------------------------------------------------------------------

typedef __attribute__((ext_vector_type(16))) __bf16 v16bf;
typedef __attribute__((ext_vector_type(8)))  __bf16 v8bf;
typedef __attribute__((ext_vector_type(8)))  float  v8f;

#define Bsz   8
#define Lsz   2048
#define Dsz   64
#define RELsz 33
#define NEGC  1e9f

static __device__ __forceinline__ unsigned short f2bf(float x) {
  return __builtin_bit_cast(unsigned short, (__bf16)x);
}
static __device__ __forceinline__ v8bf ldg8(const unsigned short* p) {
  return *(const v8bf*)p;
}
static __device__ __forceinline__ v16bf cat8(v8bf lo, v8bf hi) {
  return __builtin_shufflevector(lo, hi, 0,1,2,3,4,5,6,7,8,9,10,11,12,13,14,15);
}
static __device__ __forceinline__ v8f wmma_bf16(v16bf a, v16bf b, v8f c) {
  // v_wmma_f32_16x16x32_bf16  D = A(16x32) * B(32x16) + C
  return __builtin_amdgcn_wmma_f32_16x16x32_bf16(false, a, false, b,
                                                 (short)0, c, false, false);
}
static __device__ __forceinline__ float redmax16(float x) {
  x = fmaxf(x, __shfl_xor(x, 1, 32));
  x = fmaxf(x, __shfl_xor(x, 2, 32));
  x = fmaxf(x, __shfl_xor(x, 4, 32));
  x = fmaxf(x, __shfl_xor(x, 8, 32));
  return x;
}
static __device__ __forceinline__ float redsum16(float x) {
  x += __shfl_xor(x, 1, 32);
  x += __shfl_xor(x, 2, 32);
  x += __shfl_xor(x, 4, 32);
  x += __shfl_xor(x, 8, 32);
  return x;
}

// ------------------------- pre-pass kernels --------------------------------

__global__ void cvt_qk(const float* __restrict__ Q, const float* __restrict__ K,
                       unsigned short* __restrict__ Qbf,
                       unsigned short* __restrict__ Kbf, int n) {
  int i = blockIdx.x * blockDim.x + threadIdx.x;
  if (i < n) { Qbf[i] = f2bf(Q[i]); Kbf[i] = f2bf(K[i]); }
}

__global__ void cvt_vt(const float* __restrict__ V,
                       unsigned short* __restrict__ Vt, int n) {
  int i = blockIdx.x * blockDim.x + threadIdx.x;
  if (i < n) {
    int b = i / (Lsz * Dsz);
    int r = i - b * (Lsz * Dsz);
    int l = r / Dsz, d = r - l * Dsz;
    Vt[((size_t)b * Dsz + d) * Lsz + l] = f2bf(V[i]);
  }
}

__global__ void cvt_vrel(const float* __restrict__ vrel,
                         unsigned short* __restrict__ vrelT) {
  int i = blockIdx.x * blockDim.x + threadIdx.x;  // 64*32
  if (i < Dsz * 32) {
    int d = i >> 5, t = i & 31;
    vrelT[i] = (t <= 16) ? f2bf(vrel[t * Dsz + d]) : (unsigned short)0;
  }
}

// drel[b,q,t] = Q[b,q,:] . key_relative[t,:]
__global__ void drel_kernel(const float* __restrict__ Q,
                            const float* __restrict__ krel,
                            float* __restrict__ drel, int n) {
  int i = blockIdx.x * blockDim.x + threadIdx.x;  // B*L*33
  if (i < n) {
    int t = i % RELsz;
    int bq = i / RELsz;
    const float* q = Q + (size_t)bq * Dsz;
    const float* k = krel + (size_t)t * Dsz;
    float acc = 0.f;
#pragma unroll
    for (int d = 0; d < Dsz; ++d) acc += q[d] * k[d];
    drel[i] = acc;
  }
}

// ------------------------- main flash kernel -------------------------------

__global__ __launch_bounds__(128)
void flash_attn(const float* __restrict__ mask,
                const unsigned short* __restrict__ Qbf,
                const unsigned short* __restrict__ Kbf,
                const unsigned short* __restrict__ Vt,
                const float* __restrict__ drel,
                const unsigned short* __restrict__ vrelT,
                float* __restrict__ out) {
  __shared__ unsigned short Pbuf[4][16 * 40];  // per-wave 16x32 bf16, stride 40

  const int lane = threadIdx.x & 31;
  const int wave = threadIdx.x >> 5;
  const int wg   = blockIdx.x * 4 + wave;      // global q-tile id
  const int b    = wg >> 7;                    // / (L/16)
  const int q0   = (wg & 127) << 4;
  const int n    = lane & 15;                  // C/B column, A row
  const int g    = lane >> 4;                  // lane half-group
  unsigned short* P = Pbuf[wave];

  // ---- Q A-fragments (16x32 bf16 each): d 0..31 and d 32..63 ----
  const unsigned short* qrow = Qbf + ((size_t)(b * Lsz + q0 + n)) * Dsz;
  const v16bf qa0 = cat8(ldg8(qrow + g * 16),      ldg8(qrow + g * 16 + 8));
  const v16bf qa1 = cat8(ldg8(qrow + 32 + g * 16), ldg8(qrow + 32 + g * 16 + 8));

  // ---- value_relative B-fragments (loop-invariant, rows t=0..16, pad 0) ----
  const unsigned short* vr = vrelT + (size_t)n * 32 + g * 16;
  const v16bf rb0 = cat8(ldg8(vr),           ldg8(vr + 8));
  const v16bf rb1 = cat8(ldg8(vr + 16 * 32), ldg8(vr + 16 * 32 + 8));
  const v16bf rb2 = cat8(ldg8(vr + 32 * 32), ldg8(vr + 32 * 32 + 8));
  const v16bf rb3 = cat8(ldg8(vr + 48 * 32), ldg8(vr + 48 * 32 + 8));

  // far-past relative bias (t clipped to 32), one per owned row
  float rel_far[8];
#pragma unroll
  for (int r = 0; r < 8; ++r)
    rel_far[r] = drel[((size_t)(b * Lsz + q0 + r + 8 * g)) * RELsz + 32];

  v8f O0 = {0,0,0,0,0,0,0,0}, O1 = {0,0,0,0,0,0,0,0};
  v8f O2 = {0,0,0,0,0,0,0,0}, O3 = {0,0,0,0,0,0,0,0};
  float mrow[8], lrow[8];
#pragma unroll
  for (int r = 0; r < 8; ++r) { mrow[r] = -1e30f; lrow[r] = 0.f; }

  const int nchunks = (q0 + 47) >> 5;  // keys 0 .. q0+15 in chunks of 32
  for (int c = 0; c < nchunks; ++c) {
    const int kv = c << 5;
    const bool nearband = (kv + 47) > q0;  // chunk may touch the 17-band

    // ---- S = Q K^T : two 16x16 tiles ----
    const unsigned short* krow0 = Kbf + ((size_t)(b * Lsz + kv + n)) * Dsz + g * 16;
    const unsigned short* krow1 = krow0 + 16 * Dsz;
    v8f S0 = {0,0,0,0,0,0,0,0}, S1 = {0,0,0,0,0,0,0,0};
    v16bf kb;
    kb = cat8(ldg8(krow0),      ldg8(krow0 + 8));       S0 = wmma_bf16(qa0, kb, S0);
    kb = cat8(ldg8(krow0 + 32), ldg8(krow0 + 40));      S0 = wmma_bf16(qa1, kb, S0);
    kb = cat8(ldg8(krow1),      ldg8(krow1 + 8));       S1 = wmma_bf16(qa0, kb, S1);
    kb = cat8(ldg8(krow1 + 32), ldg8(krow1 + 40));      S1 = wmma_bf16(qa1, kb, S1);

    // ---- issue V B-fragment loads NOW; softmax below hides their latency ----
    const unsigned short* vcol = Vt + ((size_t)b * Dsz + n) * Lsz + kv + g * 16;
    const v16bf vb0 = cat8(ldg8(vcol),            ldg8(vcol + 8));
    const v16bf vb1 = cat8(ldg8(vcol + 16 * Lsz), ldg8(vcol + 16 * Lsz + 8));
    const v16bf vb2 = cat8(ldg8(vcol + 32 * Lsz), ldg8(vcol + 32 * Lsz + 8));
    const v16bf vb3 = cat8(ldg8(vcol + 48 * Lsz), ldg8(vcol + 48 * Lsz + 8));

    const float mt0 = (1.0f - mask[b * Lsz + kv + n]) * NEGC;
    const float mt1 = (1.0f - mask[b * Lsz + kv + 16 + n]) * NEGC;

    // ---- bias + masks -> scaled logits ----
    float s0[8], s1[8];
    if (!nearband) {  // every key is far past: t==32 bias, no causal cut
#pragma unroll
      for (int r = 0; r < 8; ++r) {
        s0[r] = (S0[r] + rel_far[r] - mt0) * 0.125f;
        s1[r] = (S1[r] + rel_far[r] - mt1) * 0.125f;
      }
    } else {
#pragma unroll
      for (int r = 0; r < 8; ++r) {
        const int q = q0 + r + 8 * g;
        const int k0 = kv + n, k1 = kv + 16 + n;
        if (k0 <= q) {
          int t = q - k0 + 16; t = t > 32 ? 32 : t;
          s0[r] = (S0[r] + drel[(size_t)(b * Lsz + q) * RELsz + t] - mt0) * 0.125f;
        } else s0[r] = -1e30f;
        if (k1 <= q) {
          int t = q - k1 + 16; t = t > 32 ? 32 : t;
          s1[r] = (S1[r] + drel[(size_t)(b * Lsz + q) * RELsz + t] - mt1) * 0.125f;
        } else s1[r] = -1e30f;
      }
    }

    // ---- online softmax update + stage P~ (bf16) in LDS ----
#pragma unroll
    for (int r = 0; r < 8; ++r) {
      const float mx   = redmax16(fmaxf(s0[r], s1[r]));
      const float mnew = fmaxf(mrow[r], mx);
      const float corr = __expf(mrow[r] - mnew);
      mrow[r] = mnew;
      const float p0 = __expf(s0[r] - mnew);
      const float p1 = __expf(s1[r] - mnew);
      lrow[r] = lrow[r] * corr + redsum16(p0 + p1);
      O0[r] *= corr; O1[r] *= corr; O2[r] *= corr; O3[r] *= corr;
      P[(r + 8 * g) * 40 + n]      = f2bf(p0);
      P[(r + 8 * g) * 40 + 16 + n] = f2bf(p1);
    }
    asm volatile("s_wait_dscnt 0" ::: "memory");  // LDS write -> read, same wave

    // ---- P~ as A-fragment, O += P~ * V (4 d-chunks) ----
    const unsigned short* prow = P + n * 40;
    const v16bf pa = cat8(*(const v8bf*)(prow + g * 16),
                          *(const v8bf*)(prow + g * 16 + 8));
    O0 = wmma_bf16(pa, vb0, O0);
    O1 = wmma_bf16(pa, vb1, O1);
    O2 = wmma_bf16(pa, vb2, O2);
    O3 = wmma_bf16(pa, vb3, O3);

    // ---- diagonal-band value_relative recombination ----
    // band A[m][t] = P~[q0+m][k=q0+m-16+t] for t<=16 (vrelT rows >16 are zero)
    if (nearband) {
      v16bf ba;
      const int q = q0 + n;
#pragma unroll
      for (int h = 0; h < 16; ++h) {
        const int t = g * 16 + h;
        const int k = q - 16 + t;
        unsigned short u = 0;
        if (k >= kv && k < kv + 32) u = P[n * 40 + (k - kv)];
        ba[h] = __builtin_bit_cast(__bf16, u);
      }
      O0 = wmma_bf16(ba, rb0, O0);
      O1 = wmma_bf16(ba, rb1, O1);
      O2 = wmma_bf16(ba, rb2, O2);
      O3 = wmma_bf16(ba, rb3, O3);
    }
  }

  // ---- normalize and write out (f32) ----
#pragma unroll
  for (int r = 0; r < 8; ++r) {
    const float inv = 1.0f / lrow[r];
    const size_t o = ((size_t)(b * Lsz + q0 + r + 8 * g)) * Dsz + n;
    out[o]      = O0[r] * inv;
    out[o + 16] = O1[r] * inv;
    out[o + 32] = O2[r] * inv;
    out[o + 48] = O3[r] * inv;
  }
}

// ------------------------------- launcher ----------------------------------

extern "C" void kernel_launch(void* const* d_in, const int* in_sizes, int n_in,
                              void* d_out, int out_size, void* d_ws, size_t ws_size,
                              hipStream_t stream) {
  const float* Q    = (const float*)d_in[0];
  const float* K    = (const float*)d_in[1];
  const float* V    = (const float*)d_in[2];
  const float* mask = (const float*)d_in[3];
  const float* krel = (const float*)d_in[4];
  const float* vrel = (const float*)d_in[5];
  float* out = (float*)d_out;

  const size_t nQ = (size_t)Bsz * Lsz * Dsz;  // 1,048,576
  unsigned short* Qbf   = (unsigned short*)d_ws;       // 2 MB
  unsigned short* Kbf   = Qbf + nQ;                    // 2 MB
  unsigned short* Vt    = Kbf + nQ;                    // 2 MB
  float*          drelp = (float*)(Vt + nQ);           // 8*2048*33*4 ~ 2.2 MB
  unsigned short* vrelT = (unsigned short*)(drelp + (size_t)Bsz * Lsz * RELsz);

  cvt_qk<<<(int)((nQ + 255) / 256), 256, 0, stream>>>(Q, K, Qbf, Kbf, (int)nQ);
  cvt_vt<<<(int)((nQ + 255) / 256), 256, 0, stream>>>(V, Vt, (int)nQ);
  cvt_vrel<<<(Dsz * 32 + 255) / 256, 256, 0, stream>>>(vrel, vrelT);
  const int ndr = Bsz * Lsz * RELsz;
  drel_kernel<<<(ndr + 255) / 256, 256, 0, stream>>>(Q, krel, drelp, ndr);

  // 1024 q-tiles (B * L/16), 4 waves per 128-thread block
  flash_attn<<<256, 128, 0, stream>>>(mask, Qbf, Kbf, Vt, drelp, vrelT, out);
}
